// SAGE_81028853006438
// MI455X (gfx1250) — matrix operand order repaired and checked by reading, not compiled, not run on hardware
//
#include <hip/hip_runtime.h>
#include <math.h>

// ---------------- problem constants ----------------
#define GG   500
#define HH   128
#define DHH  24
#define BB   2048
#define FANF 10
#define NN1  (BB * (FANF + 1))   // 22528
#define KGP  512                 // G padded to 512 for WMMA K
#define DHP  32                  // DH padded to 32
#define GPN  512                 // G padded to 512 as N

typedef _Float16 half_t;
typedef half_t v16h __attribute__((ext_vector_type(16)));
typedef half_t v8h  __attribute__((ext_vector_type(8)));
typedef float  v8f  __attribute__((ext_vector_type(8)));

// ---------------- reductions ----------------
__device__ __forceinline__ float block_sum(float v, float* sh) {
    int t = threadIdx.x;
    sh[t] = v;
    __syncthreads();
    for (int s = blockDim.x >> 1; s > 0; s >>= 1) {
        if (t < s) sh[t] += sh[t + s];
        __syncthreads();
    }
    float r = sh[0];
    __syncthreads();
    return r;
}

__device__ __forceinline__ float block_max(float v, float* sh) {
    int t = threadIdx.x;
    sh[t] = v;
    __syncthreads();
    for (int s = blockDim.x >> 1; s > 0; s >>= 1) {
        if (t < s) sh[t] = fmaxf(sh[t], sh[t + s]);
        __syncthreads();
    }
    float r = sh[0];
    __syncthreads();
    return r;
}

// ---------------- weight convert/transpose: W(K x N) f32 -> Wt(Np x Kp) f16 ----------------
__global__ void cvt_wt(const float* __restrict__ W, half_t* __restrict__ Wt,
                       int K, int N, int Kp, int Np) {
    int idx = blockIdx.x * blockDim.x + threadIdx.x;
    int total = Np * Kp;
    if (idx >= total) return;
    int n = idx / Kp, k = idx % Kp;
    float v = (n < N && k < K) ? W[(size_t)k * N + n] : 0.f;
    Wt[idx] = (half_t)v;
}

// ---------------- SAGE layer-0 aggregation (mean of 10 log1p rows) + self log1p ----------------
__global__ void sage0_agg(const float* __restrict__ x, const int* __restrict__ src0,
                          half_t* __restrict__ aggh, half_t* __restrict__ selfh) {
    int t = blockIdx.x;           // target node in [0, NN1)
    int c = threadIdx.x;          // 128 threads
    __shared__ int srcs[FANF];
    if (c < FANF) srcs[c] = src0[t * FANF + c];
    __syncthreads();
#pragma unroll
    for (int j = 0; j < 4; ++j) {
        int g = c + 128 * j;      // covers 0..511
        float s = 0.f, sv = 0.f;
        if (g < GG) {
#pragma unroll
            for (int k = 0; k < FANF; ++k)
                s += log1pf(x[(size_t)srcs[k] * GG + g]);
            s *= (1.f / FANF);
            sv = log1pf(x[(size_t)t * GG + g]);
        }
        aggh[(size_t)t * KGP + g]  = (half_t)s;
        selfh[(size_t)t * KGP + g] = (half_t)sv;
    }
}

// ---------------- SAGE layer-1 aggregation ----------------
__global__ void sage1_agg(const half_t* __restrict__ h1h, const int* __restrict__ src1,
                          half_t* __restrict__ agg1h) {
    int t = blockIdx.x;           // [0, BB)
    int c = threadIdx.x;          // 128
    __shared__ int srcs[FANF];
    if (c < FANF) srcs[c] = src1[t * FANF + c];
    __syncthreads();
    float s = 0.f;
#pragma unroll
    for (int k = 0; k < FANF; ++k)
        s += (float)h1h[(size_t)srcs[k] * HH + c];
    agg1h[(size_t)t * HH + c] = (half_t)(s * (1.f / FANF));
}

// ---------------- WMMA fragment loaders (CDNA5 layouts) ----------------
__device__ __forceinline__ v16h load_afrag(const half_t* p) {
    v8h lo = *(const v8h*)p;          // K run kb..kb+7
    v8h hi = *(const v8h*)(p + 16);   // K run kb+16..kb+23
    v16h r;
#pragma unroll
    for (int i = 0; i < 8; ++i) { r[i] = lo[i]; r[i + 8] = hi[i]; }
    return r;
}
__device__ __forceinline__ v16h load_bfrag(const half_t* p) {
    v8h lo = *(const v8h*)p;          // K run kb..kb+7
    v8h hi = *(const v8h*)(p + 16);   // K run kb+16..kb+23
    v16h r;
#pragma unroll
    for (int i = 0; i < 8; ++i) { r[i] = lo[i]; r[i + 8] = hi[i]; }
    return r;
}

// ---------------- generic WMMA GEMM: C = act(A1@W1t [+ A2@W2t] + bias) ----------------
// A: M x KP row-major f16; Wt: Np x KP row-major f16 (pre-transposed, zero-padded)
// 8 waves per block, each wave -> one 16x16 tile. DUAL/KP are compile-time so the
// K-loop fully unrolls into straight-line load clauses + WMMA accumulation chains.
template <bool DUAL, int KP>
__global__ __launch_bounds__(256) void wmma_gemm(
    const half_t* __restrict__ A1, const half_t* __restrict__ W1,
    const half_t* __restrict__ A2, const half_t* __restrict__ W2,
    const float* __restrict__ bias,
    float* __restrict__ Cf, half_t* __restrict__ Ch,
    int M, int N, int Np, int do_relu) {
    const int lane = threadIdx.x & 31;
    const int wid  = threadIdx.x >> 5;
    const int mb   = blockIdx.x << 4;
    const int nt   = (blockIdx.y << 3) + wid;
    if ((nt << 4) >= Np) return;            // wave-uniform guard, EXEC stays full

    const int am  = lane & 15;              // A row within tile
    const int akb = (lane >> 4) << 3;       // 0 or 8
    const int bn  = (nt << 4) + (lane & 15);// B column
    const int bkb = (lane >> 4) << 3;       // 0 or 8

    const half_t* a1p = A1 + (size_t)(mb + am) * KP + akb;
    const half_t* b1p = W1 + (size_t)bn * KP + bkb;

    v8f c1 = {0.f, 0.f, 0.f, 0.f, 0.f, 0.f, 0.f, 0.f};
    v8f c2 = {0.f, 0.f, 0.f, 0.f, 0.f, 0.f, 0.f, 0.f};

    if (DUAL) {
        const half_t* a2p = A2 + (size_t)(mb + am) * KP + akb;
        const half_t* b2p = W2 + (size_t)bn * KP + bkb;
#pragma unroll
        for (int k0 = 0; k0 < KP; k0 += 32) {
            v16h a  = load_afrag(a1p + k0);
            v16h b  = load_bfrag(b1p + k0);
            v16h a2 = load_afrag(a2p + k0);
            v16h b2 = load_bfrag(b2p + k0);
            c1 = __builtin_amdgcn_wmma_f32_16x16x32_f16(false, a, false, b, (short)0, c1,
                                                        false, false);
            c2 = __builtin_amdgcn_wmma_f32_16x16x32_f16(false, a2, false, b2, (short)0, c2,
                                                        false, false);
        }
    } else {
#pragma unroll
        for (int k0 = 0; k0 < KP; k0 += 32) {
            v16h a = load_afrag(a1p + k0);
            v16h b = load_bfrag(b1p + k0);
            c1 = __builtin_amdgcn_wmma_f32_16x16x32_f16(false, a, false, b, (short)0, c1,
                                                        false, false);
        }
    }

    const int col = (nt << 4) + (lane & 15);
    if (col >= N) return;
    const float bv = bias ? bias[col] : 0.f;
    const int mr = mb + ((lane >> 4) << 3);
#pragma unroll
    for (int r = 0; r < 8; ++r) {
        float v = c1[r] + bv;
        if (DUAL) v += c2[r];
        if (do_relu) v = fmaxf(v, 0.f);
        const int row = mr + r;
        if (Cf) Cf[(size_t)row * N + col] = v;
        if (Ch) Ch[(size_t)row * N + col] = (half_t)v;
    }
}

// ---------------- batchnorm: column stats then apply ----------------
__global__ void colstats(const float* __restrict__ X, float* __restrict__ mean,
                         float* __restrict__ rstd, int M, int N) {
    int c = blockIdx.x;
    float s = 0.f, ss = 0.f;
    for (int r = threadIdx.x; r < M; r += blockDim.x) {
        float v = X[(size_t)r * N + c];
        s += v; ss += v * v;
    }
    __shared__ float sh[256];
    s  = block_sum(s, sh);
    ss = block_sum(ss, sh);
    if (threadIdx.x == 0) {
        float m = s / M;
        float var = ss / M - m * m;
        mean[c] = m;
        rstd[c] = rsqrtf(var + 1e-3f);
    }
}

__global__ void bn_apply(const float* __restrict__ X, const float* __restrict__ mean,
                         const float* __restrict__ rstd, const float* __restrict__ g,
                         const float* __restrict__ be, float* __restrict__ Yf,
                         half_t* __restrict__ Yh, int M, int N, int Np, int do_relu) {
    int idx = blockIdx.x * blockDim.x + threadIdx.x;
    int total = M * Np;
    if (idx >= total) return;
    int r = idx / Np, c = idx % Np;
    float y = 0.f;
    if (c < N) {
        float v = X[(size_t)r * N + c];
        y = g[c] * (v - mean[c]) * rstd[c] + be[c];
        if (do_relu) y = fmaxf(y, 0.f);
        if (Yf) Yf[(size_t)r * N + c] = y;
    }
    if (Yh) Yh[(size_t)r * Np + c] = (half_t)y;   // zero-pads cols [N, Np)
}

// ---------------- reparameterize z + KL(z) ----------------
__global__ void reparam_z(const float* __restrict__ qzm, const float* __restrict__ qzvpre,
                          const float* __restrict__ eps_z, half_t* __restrict__ zh,
                          float* __restrict__ klz) {
    int b = blockIdx.x, t = threadIdx.x;  // 128 threads
    float m = qzm[(size_t)b * HH + t];
    float v = expf(qzvpre[(size_t)b * HH + t]) + 1e-4f;
    float z = m + sqrtf(v) * eps_z[(size_t)b * HH + t];
    zh[(size_t)b * HH + t] = (half_t)z;
    float kc = 0.5f * (v + m * m - logf(v) - 1.f);
    __shared__ float sh[128];
    kc = block_sum(kc, sh);
    if (t == 0) klz[b] = kc;
}

// ---------------- library + KL(l) (H->1 dot products per row) ----------------
__global__ void lib_kl(const float* __restrict__ q, const float* __restrict__ wm,
                       const float* __restrict__ bm, const float* __restrict__ wv,
                       const float* __restrict__ bv, const float* __restrict__ eps_l,
                       const float* __restrict__ lmean, const float* __restrict__ lvar,
                       float* __restrict__ lib, float* __restrict__ kll) {
    int b = blockIdx.x, t = threadIdx.x;  // 128 threads
    float qv = q[(size_t)b * HH + t];
    __shared__ float sh[128];
    float dm = block_sum(qv * wm[t], sh);
    float dv = block_sum(qv * wv[t], sh);
    if (t == 0) {
        float m  = dm + bm[0];
        float v  = expf(dv + bv[0]) + 1e-4f;
        lib[b]   = m + sqrtf(v) * eps_l[b];
        float lm = lmean[0], lv = lvar[0];
        kll[b]   = 0.5f * (logf(lv / v) + (v + (m - lm) * (m - lm)) / lv - 1.f);
    }
}

// ---------------- softmax + Poisson NLL + combine ----------------
__global__ void final_k(const float* __restrict__ logits, const float* __restrict__ x,
                        const float* __restrict__ lib, const float* __restrict__ klz,
                        const float* __restrict__ kll, float* __restrict__ out) {
    int b = blockIdx.x;
    __shared__ float sh[256];
    float mx = -3.4e38f;
    for (int g = threadIdx.x; g < GG; g += blockDim.x)
        mx = fmaxf(mx, logits[(size_t)b * GG + g]);
    mx = block_max(mx, sh);
    float se = 0.f;
    for (int g = threadIdx.x; g < GG; g += blockDim.x)
        se += expf(logits[(size_t)b * GG + g] - mx);
    se = block_sum(se, sh);
    float lse = mx + logf(se);
    float L = lib[b];
    float acc = 0.f;
    for (int g = threadIdx.x; g < GG; g += blockDim.x) {
        float lg   = logits[(size_t)b * GG + g] - lse;  // log softmax
        float xv   = x[(size_t)b * GG + g];
        float lrate = lg + L;
        float rate  = expf(lrate);
        acc += xv * lrate - rate - lgammaf(xv + 1.f);
    }
    acc = block_sum(acc, sh);
    if (threadIdx.x == 0) out[b] = -acc + klz[b] + kll[b];
}

// ---------------- host ----------------
static inline size_t align256(size_t x) { return (x + 255) & ~(size_t)255; }

extern "C" void kernel_launch(void* const* d_in, const int* in_sizes, int n_in,
                              void* d_out, int out_size, void* d_ws, size_t ws_size,
                              hipStream_t stream) {
    (void)in_sizes; (void)n_in; (void)out_size; (void)ws_size;
    const float* x     = (const float*)d_in[0];
    const int*   ei0   = (const int*)d_in[1];    // row 0 = src (E0 ints)
    const int*   ei1   = (const int*)d_in[2];    // row 0 = src (E1 ints)
    const float* lmean = (const float*)d_in[3];
    const float* lvar  = (const float*)d_in[4];
    const float* eps_l = (const float*)d_in[5];
    const float* eps_z = (const float*)d_in[6];
    const float* el_w  = (const float*)d_in[7];
    const float* el_b  = (const float*)d_in[8];
    const float* el_g  = (const float*)d_in[9];
    const float* el_be = (const float*)d_in[10];
    const float* el_wm = (const float*)d_in[11];
    const float* el_bm = (const float*)d_in[12];
    const float* el_wv = (const float*)d_in[13];
    const float* el_bv = (const float*)d_in[14];
    const float* c0_wl = (const float*)d_in[15];
    const float* c0_bl = (const float*)d_in[16];
    const float* c0_wr = (const float*)d_in[17];
    const float* c1_wl = (const float*)d_in[18];
    const float* c1_bl = (const float*)d_in[19];
    const float* c1_wr = (const float*)d_in[20];
    const float* ez_w  = (const float*)d_in[21];
    const float* ez_b  = (const float*)d_in[22];
    const float* ez_g  = (const float*)d_in[23];
    const float* ez_be = (const float*)d_in[24];
    const float* ez_wm = (const float*)d_in[25];
    const float* ez_bm = (const float*)d_in[26];
    const float* ez_wv = (const float*)d_in[27];
    const float* ez_bv = (const float*)d_in[28];
    const float* de_w  = (const float*)d_in[29];
    const float* de_b  = (const float*)d_in[30];
    const float* de_g  = (const float*)d_in[31];
    const float* de_be = (const float*)d_in[32];
    const float* de_ws = (const float*)d_in[33];
    const float* de_bs = (const float*)d_in[34];
    // d_in[35] de_wr, d_in[36] de_br: unused by the reference

    char* p = (char*)d_ws;
    auto alloc = [&](size_t bytes) -> void* { void* r = (void*)p; p += align256(bytes); return r; };

    half_t* aggh   = (half_t*)alloc((size_t)NN1 * KGP * 2);
    half_t* selfh  = (half_t*)alloc((size_t)NN1 * KGP * 2);
    half_t* h1h    = (half_t*)alloc((size_t)NN1 * HH * 2);
    half_t* agg1h  = (half_t*)alloc((size_t)BB * HH * 2);
    half_t* h2h    = (half_t*)alloc((size_t)BB * HH * 2);
    float*  qzpre  = (float*)alloc((size_t)BB * HH * 4);
    half_t* qzh    = (half_t*)alloc((size_t)BB * HH * 2);
    float*  qzm    = (float*)alloc((size_t)BB * HH * 4);
    float*  qzvpre = (float*)alloc((size_t)BB * HH * 4);
    half_t* zh     = (half_t*)alloc((size_t)BB * HH * 2);
    float*  klz    = (float*)alloc((size_t)BB * 4);
    float*  qpre   = (float*)alloc((size_t)BB * HH * 4);
    float*  qf     = (float*)alloc((size_t)BB * HH * 4);
    float*  lib    = (float*)alloc((size_t)BB * 4);
    float*  kll    = (float*)alloc((size_t)BB * 4);
    float*  pdpre  = (float*)alloc((size_t)BB * DHH * 4);
    half_t* pxh    = (half_t*)alloc((size_t)BB * DHP * 2);
    float*  logits = (float*)alloc((size_t)BB * GG * 4);
    half_t* c0wlT  = (half_t*)alloc((size_t)HH * KGP * 2);
    half_t* c0wrT  = (half_t*)alloc((size_t)HH * KGP * 2);
    half_t* elwT   = (half_t*)alloc((size_t)HH * KGP * 2);
    half_t* c1wlT  = (half_t*)alloc((size_t)HH * HH * 2);
    half_t* c1wrT  = (half_t*)alloc((size_t)HH * HH * 2);
    half_t* ezwT   = (half_t*)alloc((size_t)HH * HH * 2);
    half_t* ezwmT  = (half_t*)alloc((size_t)HH * HH * 2);
    half_t* ezwvT  = (half_t*)alloc((size_t)HH * HH * 2);
    half_t* dewT   = (half_t*)alloc((size_t)DHP * HH * 2);
    half_t* dewsT  = (half_t*)alloc((size_t)GPN * DHP * 2);
    float*  mb1 = (float*)alloc(HH * 4); float* rb1 = (float*)alloc(HH * 4);
    float*  mb2 = (float*)alloc(HH * 4); float* rb2 = (float*)alloc(HH * 4);
    float*  mb3 = (float*)alloc(DHP * 4); float* rb3 = (float*)alloc(DHP * 4);

    auto cvt = [&](const float* W, half_t* Wt, int K, int N, int Kp, int Np) {
        int total = Np * Kp;
        cvt_wt<<<dim3((total + 255) / 256), dim3(256), 0, stream>>>(W, Wt, K, N, Kp, Np);
    };

    // weight prep (all tiny, L2-resident)
    cvt(c0_wl, c0wlT, GG, HH, KGP, HH);
    cvt(c0_wr, c0wrT, GG, HH, KGP, HH);
    cvt(el_w,  elwT,  GG, HH, KGP, HH);
    cvt(c1_wl, c1wlT, HH, HH, HH, HH);
    cvt(c1_wr, c1wrT, HH, HH, HH, HH);
    cvt(ez_w,  ezwT,  HH, HH, HH, HH);
    cvt(ez_wm, ezwmT, HH, HH, HH, HH);
    cvt(ez_wv, ezwvT, HH, HH, HH, HH);
    cvt(de_w,  dewT,  HH, DHH, HH, DHP);
    cvt(de_ws, dewsT, DHH, GG, DHP, GPN);

    // SAGE layer 0: gather (bandwidth-dominant, ~500 MB) then fused dual-WMMA GEMM + relu
    sage0_agg<<<dim3(NN1), dim3(128), 0, stream>>>(x, ei0, aggh, selfh);
    wmma_gemm<true, KGP><<<dim3(NN1 / 16, 1), dim3(256), 0, stream>>>(
        aggh, c0wlT, selfh, c0wrT, c0_bl, nullptr, h1h, NN1, HH, HH, 1);

    // SAGE layer 1 (no relu)
    sage1_agg<<<dim3(BB), dim3(128), 0, stream>>>(h1h, ei1, agg1h);
    wmma_gemm<true, HH><<<dim3(BB / 16, 1), dim3(256), 0, stream>>>(
        agg1h, c1wlT, h1h, c1wrT, c1_bl, nullptr, h2h, BB, HH, HH, 0);

    // encoder z: BN(relu) then mean/var heads, reparameterize, KL(z)
    wmma_gemm<false, HH><<<dim3(BB / 16, 1), dim3(256), 0, stream>>>(
        h2h, ezwT, nullptr, nullptr, ez_b, qzpre, nullptr, BB, HH, HH, 0);
    colstats<<<dim3(HH), dim3(256), 0, stream>>>(qzpre, mb1, rb1, BB, HH);
    bn_apply<<<dim3((BB * HH + 255) / 256), dim3(256), 0, stream>>>(
        qzpre, mb1, rb1, ez_g, ez_be, nullptr, qzh, BB, HH, HH, 1);
    wmma_gemm<false, HH><<<dim3(BB / 16, 1), dim3(256), 0, stream>>>(
        qzh, ezwmT, nullptr, nullptr, ez_bm, qzm, nullptr, BB, HH, HH, 0);
    wmma_gemm<false, HH><<<dim3(BB / 16, 1), dim3(256), 0, stream>>>(
        qzh, ezwvT, nullptr, nullptr, ez_bv, qzvpre, nullptr, BB, HH, HH, 0);
    reparam_z<<<dim3(BB), dim3(128), 0, stream>>>(qzm, qzvpre, eps_z, zh, klz);

    // encoder l: reuses selfh[:B] == log1p(x[:B]) as f16 A
    wmma_gemm<false, KGP><<<dim3(BB / 16, 1), dim3(256), 0, stream>>>(
        selfh, elwT, nullptr, nullptr, el_b, qpre, nullptr, BB, HH, HH, 0);
    colstats<<<dim3(HH), dim3(256), 0, stream>>>(qpre, mb2, rb2, BB, HH);
    bn_apply<<<dim3((BB * HH + 255) / 256), dim3(256), 0, stream>>>(
        qpre, mb2, rb2, el_g, el_be, qf, nullptr, BB, HH, HH, 1);
    lib_kl<<<dim3(BB), dim3(128), 0, stream>>>(qf, el_wm, el_bm, el_wv, el_bv,
                                               eps_l, lmean, lvar, lib, kll);

    // decoder: z -> px (BN+relu) -> logits -> softmax/Poisson NLL
    wmma_gemm<false, HH><<<dim3(BB / 16, 1), dim3(256), 0, stream>>>(
        zh, dewT, nullptr, nullptr, de_b, pdpre, nullptr, BB, DHH, DHP, 0);
    colstats<<<dim3(DHH), dim3(256), 0, stream>>>(pdpre, mb3, rb3, BB, DHH);
    bn_apply<<<dim3((BB * DHP + 255) / 256), dim3(256), 0, stream>>>(
        pdpre, mb3, rb3, de_g, de_be, nullptr, pxh, BB, DHH, DHP, 1);
    wmma_gemm<false, DHP><<<dim3(BB / 16, (GPN + 127) / 128), dim3(256), 0, stream>>>(
        pxh, dewsT, nullptr, nullptr, de_bs, logits, nullptr, BB, GG, GPN, 0);

    final_k<<<dim3(BB), dim3(256), 0, stream>>>(logits, x, lib, klz, kll, (float*)d_out);
}
